// Intra_aggr_module_36395552866671
// MI455X (gfx1250) — matrix operand order repaired and checked by reading, not compiled, not run on hardware
//
#include <hip/hip_runtime.h>
#include <hip/hip_bf16.h>

#define NN 50000
#define EE 800000
#define DD 128

typedef __attribute__((ext_vector_type(16))) __bf16 v16bf;
typedef __attribute__((ext_vector_type(8)))  __bf16 v8bf;
typedef __attribute__((ext_vector_type(8)))  float  v8f;

struct Ptr8 { const float* p[8]; };

__device__ __forceinline__ __bf16 f2bf(float f) {
    unsigned u = __float_as_uint(f);
    u += 0x7FFFu + ((u >> 16) & 1u);          // round-to-nearest-even
    unsigned short h = (unsigned short)(u >> 16);
    __bf16 r;
    __builtin_memcpy(&r, &h, 2);
    return r;
}

// Monotone order-preserving float<->uint encoding for atomicMax-based segment max
__device__ __forceinline__ unsigned encf(float f) {
    unsigned u = __float_as_uint(f);
    return (u & 0x80000000u) ? ~u : (u | 0x80000000u);
}
__device__ __forceinline__ float decf(unsigned u) {
    return (u & 0x80000000u) ? __uint_as_float(u & 0x7FFFFFFFu) : __uint_as_float(~u);
}

// ---------------- weight conversion: 8x [128x128] f32 row-major -> bf16 col-major
__global__ __launch_bounds__(256) void conv_weights(Ptr8 wp, __bf16* __restrict__ out) {
    int idx = blockIdx.x * 256 + threadIdx.x;      // 8 * 16384 total
    if (idx >= 8 * 16384) return;
    int mat = idx >> 14;
    int rem = idx & 16383;
    int n = rem >> 7;
    int k = rem & 127;
    out[(size_t)mat * 16384 + (size_t)n * 128 + k] = f2bf(wp.p[mat][(size_t)k * 128 + n]);
}

// ---------------- per-layer init (segment max / denom / norm stats)
__global__ __launch_bounds__(256) void init_layer(unsigned* __restrict__ menc,
                                                  float* __restrict__ denom,
                                                  float* __restrict__ stats) {
    int i = blockIdx.x * 256 + threadIdx.x;
    if (i < NN) { menc[i] = 0x007FFFFFu; /* encf(-inf) */ denom[i] = 0.0f; }
    if (i < 2) stats[i] = 0.0f;
}

// ---------------- fused 4-way GEMM: q/k/v/s = x @ W + b   (WMMA bf16)
__global__ __launch_bounds__(128) void gemm_qkvs(
    const float* __restrict__ xin, const __bf16* __restrict__ Wt,
    const float* __restrict__ bq, const float* __restrict__ bk,
    const float* __restrict__ bv, const float* __restrict__ bs,
    float* __restrict__ q, float* __restrict__ k,
    float* __restrict__ v, float* __restrict__ s)
{
    __shared__ __bf16 xs[16][136];                 // padded to dodge bank conflicts
    const int tid  = threadIdx.x;
    const int row0 = blockIdx.x * 16;

    #pragma unroll
    for (int i = tid; i < 16 * 128; i += 128) {
        int r = i >> 7, c = i & 127;
        xs[r][c] = f2bf(xin[(size_t)(row0 + r) * 128 + c]);
    }
    __syncthreads();

    const int wave = tid >> 5;
    const int lane = tid & 31;
    const int mn   = lane & 15;                    // row (A) / col (B,D)
    const int g    = lane >> 4;

    // A fragments (16-bit A 16x32 layout): lane-group g, element i -> K=16g+i (contiguous, matches B)
    v16bf afr[4];
    #pragma unroll
    for (int kt = 0; kt < 4; ++kt) {
        const v8bf* ap = (const v8bf*)&xs[mn][kt * 32 + 16 * g];
        v8bf lo = ap[0];
        v8bf hi = ap[1];                           // +8 elements
        #pragma unroll
        for (int i = 0; i < 8; ++i) { afr[kt][i] = lo[i]; afr[kt][8 + i] = hi[i]; }
    }

    const __bf16* W    = Wt + (size_t)wave * (128 * 128);
    const float*  bias = (wave == 0) ? bq : (wave == 1) ? bk : (wave == 2) ? bv : bs;
    float*        out  = (wave == 0) ? q  : (wave == 1) ? k  : (wave == 2) ? v  : s;

    #pragma unroll
    for (int nt = 0; nt < 8; ++nt) {
        v8f acc = {};
        const __bf16* wrow = W + (size_t)(nt * 16 + mn) * 128;
        #pragma unroll
        for (int kt = 0; kt < 4; ++kt) {
            // B 32x16 layout: element i <-> K = 32*kt + 16*g + i (contiguous 32B load)
            v16bf bfr = *(const v16bf*)(wrow + kt * 32 + 16 * g);
            acc = __builtin_amdgcn_wmma_f32_16x16x32_bf16(
                false, afr[kt], false, bfr, (short)0, acc, false, false);
        }
        float bb = bias[nt * 16 + mn];
        #pragma unroll
        for (int r = 0; r < 8; ++r)                // D: VGPR r -> row r + 8g, col = nt*16+mn
            out[(size_t)(row0 + r + 8 * g) * 128 + nt * 16 + mn] = acc[r] + bb;
    }
}

// ---------------- edge score + segment max (wave per edge)
__global__ __launch_bounds__(256) void edge_score(
    const float* __restrict__ q, const float* __restrict__ k,
    const long long* __restrict__ ei, float* __restrict__ score,
    unsigned* __restrict__ menc)
{
    int e    = blockIdx.x * 8 + (threadIdx.x >> 5);
    int lane = threadIdx.x & 31;
    long long s = ei[e];
    long long d = ei[EE + e];
    float4 qa = ((const float4*)(q + (size_t)d * 128))[lane];
    float4 ka = ((const float4*)(k + (size_t)s * 128))[lane];
    float p = qa.x * ka.x + qa.y * ka.y + qa.z * ka.z + qa.w * ka.w;
    #pragma unroll
    for (int off = 16; off > 0; off >>= 1) p += __shfl_xor(p, off, 32);
    if (lane == 0) {
        float sc = p * 0.08838834764831845f;       // 1/sqrt(128)
        score[e] = sc;
        atomicMax(&menc[d], encf(sc));
    }
}

// ---------------- exp(score - m[dst]) and segment sum
__global__ __launch_bounds__(256) void edge_exp(
    const long long* __restrict__ ei, const unsigned* __restrict__ menc,
    float* __restrict__ score, float* __restrict__ denom)
{
    int e = blockIdx.x * 256 + threadIdx.x;
    long long d = ei[EE + e];
    float ev = __expf(score[e] - decf(menc[d]));
    score[e] = ev;
    atomicAdd(&denom[d], ev);
}

// ---------------- agg[dst] += alpha * v[src]  (wave per edge)
__global__ __launch_bounds__(256) void edge_agg(
    const float* __restrict__ v, const float* __restrict__ score,
    const float* __restrict__ denom, const long long* __restrict__ ei,
    float* __restrict__ agg)
{
    int e    = blockIdx.x * 8 + (threadIdx.x >> 5);
    int lane = threadIdx.x & 31;
    long long s = ei[e];
    long long d = ei[EE + e];
    float alpha = score[e] / denom[d];
    float4 vv = ((const float4*)(v + (size_t)s * 128))[lane];
    float* ag = agg + (size_t)d * 128 + lane * 4;
    atomicAdd(ag + 0, alpha * vv.x);
    atomicAdd(ag + 1, alpha * vv.y);
    atomicAdd(ag + 2, alpha * vv.z);
    atomicAdd(ag + 3, alpha * vv.w);
}

// ---------------- global sum / sumsq for graph norm
__global__ __launch_bounds__(256) void stats_kernel(const float* __restrict__ agg,
                                                    float* __restrict__ stats)
{
    __shared__ float s0[256], s1[256];
    float a = 0.f, b = 0.f;
    for (size_t i = (size_t)blockIdx.x * 256 + threadIdx.x; i < (size_t)NN * 128;
         i += (size_t)gridDim.x * 256) {
        float x = agg[i];
        a += x;
        b += x * x;
    }
    s0[threadIdx.x] = a; s1[threadIdx.x] = b;
    __syncthreads();
    for (int st = 128; st > 0; st >>= 1) {
        if (threadIdx.x < st) {
            s0[threadIdx.x] += s0[threadIdx.x + st];
            s1[threadIdx.x] += s1[threadIdx.x + st];
        }
        __syncthreads();
    }
    if (threadIdx.x == 0) { atomicAdd(&stats[0], s0[0]); atomicAdd(&stats[1], s1[0]); }
}

// ---------------- graph-norm + ReLU + residual
__global__ __launch_bounds__(256) void norm_residual(
    const float* __restrict__ agg, const float* __restrict__ stats,
    const float* __restrict__ gw, const float* __restrict__ gb,
    const float* __restrict__ xin, float* __restrict__ xout)
{
    size_t i = (size_t)blockIdx.x * 256 + threadIdx.x;
    int c = (int)(i & 127);
    const float invND = 1.0f / ((float)NN * 128.0f);
    float mu  = stats[0] * invND;
    float var = stats[1] * invND - mu * mu;
    float r   = rsqrtf(var + 1e-5f);
    float val = (agg[i] - mu) * r * gw[c] + gb[c];
    xout[i] = fmaxf(val, 0.0f) + xin[i];
}

extern "C" void kernel_launch(void* const* d_in, const int* in_sizes, int n_in,
                              void* d_out, int out_size, void* d_ws, size_t ws_size,
                              hipStream_t stream)
{
    (void)in_sizes; (void)n_in; (void)out_size; (void)ws_size;

    const float*     x  = (const float*)d_in[0];
    const long long* ei = (const long long*)d_in[1];

    Ptr8 wp;
    const float* bias[2][4];
    const float* gw[2];
    const float* gb[2];
    for (int l = 0; l < 2; ++l) {
        for (int j = 0; j < 4; ++j) {
            wp.p[l * 4 + j] = (const float*)d_in[2 + l * 10 + j];
            bias[l][j]      = (const float*)d_in[6 + l * 10 + j];
        }
        gw[l] = (const float*)d_in[10 + l * 10];
        gb[l] = (const float*)d_in[11 + l * 10];
    }

    char* ws = (char*)d_ws;
    size_t off = 0;
    __bf16* Wt = (__bf16*)(ws + off); off += (size_t)8 * 16384 * 2;   // 256 KB
    float* q     = (float*)(ws + off); off += (size_t)NN * 128 * 4;
    float* kbuf  = (float*)(ws + off); off += (size_t)NN * 128 * 4;
    float* vbuf  = (float*)(ws + off); off += (size_t)NN * 128 * 4;
    float* agg   = (float*)(ws + off); off += (size_t)NN * 128 * 4;
    float* x0    = (float*)(ws + off); off += (size_t)NN * 128 * 4;
    float* score = (float*)(ws + off); off += (size_t)EE * 4;
    unsigned* menc = (unsigned*)(ws + off); off += (size_t)NN * 4;
    float* denom = (float*)(ws + off); off += (size_t)NN * 4;
    float* stats = (float*)(ws + off); off += 64;

    conv_weights<<<(8 * 16384 + 255) / 256, 256, 0, stream>>>(wp, Wt);

    for (int l = 0; l < 2; ++l) {
        const float* xin  = l ? x0 : x;
        float*       xout = l ? (float*)d_out : x0;
        const __bf16* Wl  = Wt + (size_t)l * 4 * 16384;

        init_layer<<<(NN + 255) / 256, 256, 0, stream>>>(menc, denom, stats);
        gemm_qkvs<<<NN / 16, 128, 0, stream>>>(xin, Wl,
            bias[l][0], bias[l][1], bias[l][2], bias[l][3], q, kbuf, vbuf, agg);
        edge_score<<<EE / 8, 256, 0, stream>>>(q, kbuf, ei, score, menc);
        edge_exp<<<EE / 256, 256, 0, stream>>>(ei, menc, score, denom);
        edge_agg<<<EE / 8, 256, 0, stream>>>(vbuf, score, denom, ei, agg);
        stats_kernel<<<1024, 256, 0, stream>>>(agg, stats);
        norm_residual<<<(NN * 128) / 256, 256, 0, stream>>>(agg, stats, gw[l], gb[l], xin, xout);
    }
}
